// MultiScaleBlock_67267777790171
// MI455X (gfx1250) — compile-verified, matrix operand order checked by reading, not run on hardware
//
#include <hip/hip_runtime.h>
#include <hip/hip_fp16.h>

typedef __attribute__((ext_vector_type(16))) _Float16 v16h;
typedef __attribute__((ext_vector_type(8)))  _Float16 v8h;
typedef __attribute__((ext_vector_type(8)))  float    v8f;

#define BQ     4
#define NTOK   6273
#define DIMC   768
#define HEADS  8
#define HDIM   96
#define HIDC   3072
#define NQ     1569          // pooled tokens (incl cls)
#define NQP    1600          // padded pooled tokens
#define NTOKF  (BQ*NTOK)     // 25092
#define NTOKP  25104         // padded to multiple of 16
#define TT 8
#define HH 28
#define WW 28
#define OH 14
#define OW 14
#define ATT_SCALE 0.10206207261596577f   // 96^-0.5

// ---------------------------------------------------------------------------
// Fragment loader: row-major f16 matrix, leading dim `ld` (halfs).
// A-matrix 16x32 f16 layout (and symmetric B layout):
//   lane<16 : row = lane,    K = {k0+0..7, k0+16..23}
//   lane>=16: row = lane-16, K = {k0+8..15, k0+24..31}
// ---------------------------------------------------------------------------
__device__ inline v16h load_frag_rm(const _Float16* __restrict__ base, int ld,
                                    int row0, int k0, int lane) {
  int r  = lane & 15;
  int hh = lane >> 4;
  const _Float16* p = base + (size_t)(row0 + r) * ld + k0 + hh * 8;
  v8h lo = *(const v8h*)p;
  v8h hi = *(const v8h*)(p + 16);
  v16h out;
#pragma unroll
  for (int i = 0; i < 8; ++i) { out[i] = lo[i]; out[i + 8] = hi[i]; }
  return out;
}

__device__ inline v8f wmma16(v16h a, v16h b, v8f c) {
  return __builtin_amdgcn_wmma_f32_16x16x32_f16(false, a, false, b, (short)0, c,
                                                false, false);
}

// ---------------------------------------------------------------------------
// f32 -> f16 conversion (weights)
// ---------------------------------------------------------------------------
__global__ void cvt_kernel(const float* __restrict__ in, _Float16* __restrict__ out, int n) {
  int i = blockIdx.x * 256 + threadIdx.x;
  if (i < n) out[i] = (_Float16)in[i];
}

// ---------------------------------------------------------------------------
// LayerNorm over 768, one wave per row, f16 output; invalid rows -> zeros
// ---------------------------------------------------------------------------
__global__ __launch_bounds__(128) void ln_kernel(const float* __restrict__ in,
    const float* __restrict__ w, const float* __restrict__ b,
    _Float16* __restrict__ out, int perB, int valid) {
  int lane = threadIdx.x & 31, wid = threadIdx.x >> 5;
  int row = blockIdx.x * 4 + wid;
  _Float16* orow = out + (size_t)row * DIMC;
  if ((row % perB) >= valid) {
    for (int j = lane; j < DIMC; j += 32) orow[j] = (_Float16)0.f;
    return;
  }
  const float* irow = in + (size_t)row * DIMC;
  float vr[24];
  float s = 0.f, sq = 0.f;
#pragma unroll
  for (int j = 0; j < 24; ++j) {
    float v = irow[j * 32 + lane];
    vr[j] = v; s += v; sq += v * v;
  }
#pragma unroll
  for (int m = 1; m < 32; m <<= 1) { s += __shfl_xor(s, m, 32); sq += __shfl_xor(sq, m, 32); }
  float mu = s * (1.f / DIMC);
  float var = sq * (1.f / DIMC) - mu * mu;
  float rstd = rsqrtf(var + 1e-5f);
#pragma unroll
  for (int j = 0; j < 24; ++j) {
    int c = j * 32 + lane;
    orow[c] = (_Float16)((vr[j] - mu) * rstd * w[c] + b[c]);
  }
}

// ---------------------------------------------------------------------------
// Generic WMMA GEMM: C[M,N] = A[M,K] * W[N,K]^T + bias
//   block = 128 threads = 4 waves, wave tile = 16(M) x 64(N)
// ---------------------------------------------------------------------------
template<bool GELU_, bool FINAL_>
__global__ __launch_bounds__(128) void gemm_kernel(
    const _Float16* __restrict__ A, const _Float16* __restrict__ W,
    const float* __restrict__ bias, float* __restrict__ Cf,
    _Float16* __restrict__ Ch, const float* __restrict__ resid,
    int M, int N, int K) {
  const int lane = threadIdx.x & 31;
  const int wid  = threadIdx.x >> 5;
  const int n0   = (blockIdx.x * 4 + wid) * 64;
  const int row0 = blockIdx.y * 16;
  if (n0 >= N) return;

  v8f acc[4] = {};
  for (int k0 = 0; k0 < K; k0 += 32) {
    if (k0 + 32 < K)
      __builtin_prefetch(A + (size_t)(row0 + (lane & 15)) * K + k0 + 32, 0, 0);
    v16h af = load_frag_rm(A, K, row0, k0, lane);
    v16h b0 = load_frag_rm(W, K, n0,      k0, lane);
    v16h b1 = load_frag_rm(W, K, n0 + 16, k0, lane);
    v16h b2 = load_frag_rm(W, K, n0 + 32, k0, lane);
    v16h b3 = load_frag_rm(W, K, n0 + 48, k0, lane);
    acc[0] = wmma16(af, b0, acc[0]);
    acc[1] = wmma16(af, b1, acc[1]);
    acc[2] = wmma16(af, b2, acc[2]);
    acc[3] = wmma16(af, b3, acc[3]);
  }

  const int mb = (lane >> 4) * 8;
#pragma unroll
  for (int j = 0; j < 4; ++j) {
    int n = n0 + j * 16 + (lane & 15);
#pragma unroll
    for (int i = 0; i < 8; ++i) {
      int row = row0 + mb + i;
      float v = acc[j][i] + bias[n];
      if (GELU_) v = 0.5f * v * (1.f + erff(v * 0.7071067811865476f));
      if (FINAL_) {
        int bidx = row / NQP;
        int t    = row - bidx * NQP;
        if (t < NQ) {
          v += resid[(size_t)row * N + n];
          Cf[((size_t)(bidx * NQ + t)) * N + n] = v;
        }
      } else {
        if (Cf) Cf[(size_t)row * N + n] = v;
        if (Ch) Ch[(size_t)row * N + n] = (_Float16)v;
      }
    }
  }
}

// ---------------------------------------------------------------------------
// Depthwise 3x3x3 conv pool (stride 1,2,2, pad 1) + per-head LayerNorm.
// One wave per pooled token (lane handles d = lane, lane+32, lane+64).
// mode: 0 = q (SCALE folded, [bh][tok][d]), 1 = k ([bh][tok][d]),
//       2 = v transposed ([bh][d][tok]).
// ---------------------------------------------------------------------------
__global__ __launch_bounds__(128) void attn_pool_kernel(
    const float* __restrict__ src, const float* __restrict__ cw,
    const float* __restrict__ nw, const float* __restrict__ nb,
    _Float16* __restrict__ dst, int mode) {
  int lane = threadIdx.x & 31, wid = threadIdx.x >> 5;
  int tk = blockIdx.x * 4 + wid;              // 0..1599
  int bh = blockIdx.y;
  int b = bh >> 3, h = bh & 7;

  if (tk >= NQ) {                              // zero padding tokens
#pragma unroll
    for (int j = 0; j < 3; ++j) {
      int d = lane + 32 * j;
      if (mode == 2) dst[((size_t)bh * HDIM + d) * NQP + tk] = (_Float16)0.f;
      else           dst[((size_t)bh * NQP + tk) * HDIM + d] = (_Float16)0.f;
    }
    return;
  }

  float vals[3];
  if (tk == 0) {                               // cls token: no conv
    const float* r = src + (size_t)(b * NTOK) * DIMC + h * HDIM;
#pragma unroll
    for (int j = 0; j < 3; ++j) vals[j] = r[lane + 32 * j];
  } else {
    int s = tk - 1;
    int t = s / (OH * OW);
    int rem = s - t * (OH * OW);
    int oy = rem / OW, ox = rem - (rem / OW) * OW;
    vals[0] = vals[1] = vals[2] = 0.f;
    for (int kz = 0; kz < 3; ++kz) {
      int z = t + kz - 1; if (z < 0 || z >= TT) continue;
      for (int ky = 0; ky < 3; ++ky) {
        int y = 2 * oy + ky - 1; if (y < 0 || y >= HH) continue;
        for (int kx = 0; kx < 3; ++kx) {
          int xx = 2 * ox + kx - 1; if (xx < 0 || xx >= WW) continue;
          int n = 1 + z * (HH * WW) + y * WW + xx;
          const float* r = src + (size_t)(b * NTOK + n) * DIMC + h * HDIM;
          int widx = kz * 9 + ky * 3 + kx;
#pragma unroll
          for (int j = 0; j < 3; ++j) {
            int d = lane + 32 * j;
            vals[j] += r[d] * cw[d * 27 + widx];
          }
        }
      }
    }
  }
  float s1 = vals[0] + vals[1] + vals[2];
  float s2 = vals[0] * vals[0] + vals[1] * vals[1] + vals[2] * vals[2];
#pragma unroll
  for (int m = 1; m < 32; m <<= 1) { s1 += __shfl_xor(s1, m, 32); s2 += __shfl_xor(s2, m, 32); }
  float mu = s1 * (1.f / HDIM);
  float var = s2 * (1.f / HDIM) - mu * mu;
  float rstd = rsqrtf(var + 1e-5f);
  float scl = (mode == 0) ? ATT_SCALE : 1.f;
#pragma unroll
  for (int j = 0; j < 3; ++j) {
    int d = lane + 32 * j;
    float o = ((vals[j] - mu) * rstd * nw[d] + nb[d]) * scl;
    if (mode == 2) dst[((size_t)bh * HDIM + d) * NQP + tk] = (_Float16)o;
    else           dst[((size_t)bh * NQP + tk) * HDIM + d] = (_Float16)o;
  }
}

// ---------------------------------------------------------------------------
// Flash-style attention. Wave per 16-query tile; 50 chunks of 32 keys.
// S = Q*K^T via 6 WMMA, online softmax, P through LDS (C->A layout),
// O += P*V via 6 WMMA. Writes f16 into [b][tok][h*96+d] for proj GEMM.
// ---------------------------------------------------------------------------
__global__ __launch_bounds__(128) void attn_kernel(
    const _Float16* __restrict__ qp, const _Float16* __restrict__ kp,
    const _Float16* __restrict__ vT, _Float16* __restrict__ o16) {
  __shared__ __align__(16) _Float16 Pl[4][16 * 32];
  const int lane = threadIdx.x & 31, wid = threadIdx.x >> 5;
  const int bh = blockIdx.y;
  const int b = bh >> 3, h = bh & 7;
  const int qt = blockIdx.x * 4 + wid;        // 0..99
  const _Float16* Q  = qp + (size_t)bh * NQP * HDIM;
  const _Float16* Kp = kp + (size_t)bh * NQP * HDIM;
  const _Float16* Vt = vT + (size_t)bh * HDIM * NQP;

  v16h qa0 = load_frag_rm(Q, HDIM, qt * 16, 0,  lane);
  v16h qa1 = load_frag_rm(Q, HDIM, qt * 16, 32, lane);
  v16h qa2 = load_frag_rm(Q, HDIM, qt * 16, 64, lane);

  float mrow[8], lrow[8];
  v8f oacc[6] = {};
#pragma unroll
  for (int i = 0; i < 8; ++i) { mrow[i] = -1e30f; lrow[i] = 0.f; }
  const int mb = (lane >> 4) * 8;
  _Float16* myP = Pl[wid];

  for (int kc = 0; kc < NQP; kc += 32) {
    v8f s0 = {}, s1 = {};
    {
      v16h kb0 = load_frag_rm(Kp, HDIM, kc,      0,  lane);
      v16h kb1 = load_frag_rm(Kp, HDIM, kc + 16, 0,  lane);
      s0 = wmma16(qa0, kb0, s0); s1 = wmma16(qa0, kb1, s1);
      kb0 = load_frag_rm(Kp, HDIM, kc,      32, lane);
      kb1 = load_frag_rm(Kp, HDIM, kc + 16, 32, lane);
      s0 = wmma16(qa1, kb0, s0); s1 = wmma16(qa1, kb1, s1);
      kb0 = load_frag_rm(Kp, HDIM, kc,      64, lane);
      kb1 = load_frag_rm(Kp, HDIM, kc + 16, 64, lane);
      s0 = wmma16(qa2, kb0, s0); s1 = wmma16(qa2, kb1, s1);
    }
    int n0 = kc + (lane & 15), n1 = n0 + 16;
    bool vld0 = n0 < NQ, vld1 = n1 < NQ;
#pragma unroll
    for (int i = 0; i < 8; ++i) {
      float a = vld0 ? s0[i] : -1e30f;
      float c = vld1 ? s1[i] : -1e30f;
      float rm = fmaxf(a, c);
      rm = fmaxf(rm, __shfl_xor(rm, 1, 32));
      rm = fmaxf(rm, __shfl_xor(rm, 2, 32));
      rm = fmaxf(rm, __shfl_xor(rm, 4, 32));
      rm = fmaxf(rm, __shfl_xor(rm, 8, 32));
      float mnew = fmaxf(mrow[i], rm);
      float p0 = __expf(a - mnew);
      float p1 = __expf(c - mnew);
      float rs = p0 + p1;
      rs += __shfl_xor(rs, 1, 32); rs += __shfl_xor(rs, 2, 32);
      rs += __shfl_xor(rs, 4, 32); rs += __shfl_xor(rs, 8, 32);
      float corr = __expf(mrow[i] - mnew);
      lrow[i] = lrow[i] * corr + rs;
      mrow[i] = mnew;
#pragma unroll
      for (int t = 0; t < 6; ++t) oacc[t][i] *= corr;
      myP[(mb + i) * 32 + (lane & 15)]      = (_Float16)p0;
      myP[(mb + i) * 32 + (lane & 15) + 16] = (_Float16)p1;
    }
    __syncthreads();
    v16h pa;
    {
      int r = lane & 15, hhf = lane >> 4;
      const _Float16* pp = myP + r * 32 + hhf * 8;
      v8h lo = *(const v8h*)pp;
      v8h hi = *(const v8h*)(pp + 16);
#pragma unroll
      for (int i = 0; i < 8; ++i) { pa[i] = lo[i]; pa[i + 8] = hi[i]; }
    }
#pragma unroll
    for (int t = 0; t < 6; ++t) {
      v16h vb = load_frag_rm(Vt, NQP, t * 16, kc, lane);
      oacc[t] = wmma16(pa, vb, oacc[t]);
    }
    __syncthreads();
  }

#pragma unroll
  for (int i = 0; i < 8; ++i) {
    float inv = 1.f / lrow[i];
    size_t row = (size_t)b * NQP + qt * 16 + mb + i;
#pragma unroll
    for (int t = 0; t < 6; ++t)
      o16[row * DIMC + h * HDIM + t * 16 + (lane & 15)] = (_Float16)(oacc[t][i] * inv);
  }
}

// ---------------------------------------------------------------------------
// max_pool_skip(x) + proj output -> x2  (f32, per-b padded to NQP rows)
// ---------------------------------------------------------------------------
__global__ void pool_add_kernel(const float* __restrict__ x,
                                const float* __restrict__ o2,
                                float* __restrict__ x2) {
  size_t id = (size_t)blockIdx.x * 256 + threadIdx.x;
  const size_t total = (size_t)BQ * NQ * DIMC;
  if (id >= total) return;
  int c = (int)(id % DIMC);
  size_t r = id / DIMC;
  int n = (int)(r % NQ);
  int b = (int)(r / NQ);
  float m;
  if (n == 0) {
    m = x[(size_t)(b * NTOK) * DIMC + c];
  } else {
    int s = n - 1;
    int t = s / (OH * OW);
    int rem = s - t * (OH * OW);
    int oy = rem / OW, ox = rem - (rem / OW) * OW;
    m = -INFINITY;
    for (int ky = 0; ky < 3; ++ky) {
      int y = 2 * oy + ky - 1; if (y < 0 || y >= HH) continue;
      for (int kx = 0; kx < 3; ++kx) {
        int xx = 2 * ox + kx - 1; if (xx < 0 || xx >= WW) continue;
        m = fmaxf(m, x[(size_t)(b * NTOK + 1 + t * (HH * WW) + y * WW + xx) * DIMC + c]);
      }
    }
  }
  size_t orow = (size_t)b * NQP + n;
  x2[orow * DIMC + c] = m + o2[orow * DIMC + c];
}

// ---------------------------------------------------------------------------
extern "C" void kernel_launch(void* const* d_in, const int* in_sizes, int n_in,
                              void* d_out, int out_size, void* d_ws, size_t ws_size,
                              hipStream_t stream) {
  const float* x      = (const float*)d_in[0];
  const float* n1w    = (const float*)d_in[1];
  const float* n1b    = (const float*)d_in[2];
  const float* wq     = (const float*)d_in[3];
  const float* bq     = (const float*)d_in[4];
  const float* wk     = (const float*)d_in[5];
  const float* bk     = (const float*)d_in[6];
  const float* wv     = (const float*)d_in[7];
  const float* bv     = (const float*)d_in[8];
  const float* pqw    = (const float*)d_in[9];
  const float* pkw    = (const float*)d_in[10];
  const float* pvw    = (const float*)d_in[11];
  const float* nqw    = (const float*)d_in[12];
  const float* nqb    = (const float*)d_in[13];
  const float* nkw    = (const float*)d_in[14];
  const float* nkb    = (const float*)d_in[15];
  const float* nvw    = (const float*)d_in[16];
  const float* nvb    = (const float*)d_in[17];
  const float* projw  = (const float*)d_in[18];
  const float* projb  = (const float*)d_in[19];
  const float* n2w    = (const float*)d_in[20];
  const float* n2b    = (const float*)d_in[21];
  const float* fc1w   = (const float*)d_in[22];
  const float* fc1b   = (const float*)d_in[23];
  const float* fc2w   = (const float*)d_in[24];
  const float* fc2b   = (const float*)d_in[25];
  float* out = (float*)d_out;

  // ---- carve workspace ----
  char* p = (char*)d_ws;
  auto carve = [&](size_t bytes) -> char* {
    char* r = p; p += (bytes + 255) & ~(size_t)255; return r;
  };
  _Float16* wq16  = (_Float16*)carve((size_t)DIMC * DIMC * 2);
  _Float16* wk16  = (_Float16*)carve((size_t)DIMC * DIMC * 2);
  _Float16* wv16  = (_Float16*)carve((size_t)DIMC * DIMC * 2);
  _Float16* wp16  = (_Float16*)carve((size_t)DIMC * DIMC * 2);
  _Float16* w116  = (_Float16*)carve((size_t)HIDC * DIMC * 2);
  _Float16* w216  = (_Float16*)carve((size_t)DIMC * HIDC * 2);
  _Float16* xn    = (_Float16*)carve((size_t)NTOKP * DIMC * 2);
  float*    qf    = (float*)   carve((size_t)NTOKP * DIMC * 4);
  float*    kf    = (float*)   carve((size_t)NTOKP * DIMC * 4);
  float*    vf    = (float*)   carve((size_t)NTOKP * DIMC * 4);
  _Float16* qpool = (_Float16*)carve((size_t)BQ * HEADS * NQP * HDIM * 2);
  _Float16* kpool = (_Float16*)carve((size_t)BQ * HEADS * NQP * HDIM * 2);
  _Float16* vTp   = (_Float16*)carve((size_t)BQ * HEADS * HDIM * NQP * 2);
  _Float16* o16   = (_Float16*)carve((size_t)BQ * NQP * DIMC * 2);
  float*    o2    = (float*)   carve((size_t)BQ * NQP * DIMC * 4);
  float*    x2    = (float*)   carve((size_t)BQ * NQP * DIMC * 4);
  _Float16* xn2   = (_Float16*)carve((size_t)BQ * NQP * DIMC * 2);
  _Float16* hbuf  = (_Float16*)carve((size_t)BQ * NQP * HIDC * 2);
  (void)ws_size; (void)n_in; (void)in_sizes; (void)out_size;

  // ---- weight conversion to f16 ----
  auto cvt = [&](const float* src, _Float16* dst, int n) {
    cvt_kernel<<<dim3((n + 255) / 256), dim3(256), 0, stream>>>(src, dst, n);
  };
  cvt(wq,   wq16, DIMC * DIMC);
  cvt(wk,   wk16, DIMC * DIMC);
  cvt(wv,   wv16, DIMC * DIMC);
  cvt(projw, wp16, DIMC * DIMC);
  cvt(fc1w, w116, HIDC * DIMC);
  cvt(fc2w, w216, DIMC * HIDC);

  // ---- LN1: x -> xn (f16, padded rows zeroed) ----
  ln_kernel<<<dim3(NTOKP / 4), dim3(128), 0, stream>>>(x, n1w, n1b, xn, NTOKP, NTOKF);

  // ---- QKV projections (WMMA) ----
  dim3 gq(DIMC / 256, NTOKP / 16);
  gemm_kernel<false, false><<<gq, 128, 0, stream>>>(xn, wq16, bq, qf, ((_Float16*)nullptr), nullptr, NTOKP, DIMC, DIMC);
  gemm_kernel<false, false><<<gq, 128, 0, stream>>>(xn, wk16, bk, kf, ((_Float16*)nullptr), nullptr, NTOKP, DIMC, DIMC);
  gemm_kernel<false, false><<<gq, 128, 0, stream>>>(xn, wv16, bv, vf, ((_Float16*)nullptr), nullptr, NTOKP, DIMC, DIMC);

  // ---- pooling conv + per-head LN ----
  dim3 gp(NQP / 4, BQ * HEADS);
  attn_pool_kernel<<<gp, 128, 0, stream>>>(qf, pqw, nqw, nqb, qpool, 0);
  attn_pool_kernel<<<gp, 128, 0, stream>>>(kf, pkw, nkw, nkb, kpool, 1);
  attn_pool_kernel<<<gp, 128, 0, stream>>>(vf, pvw, nvw, nvb, vTp,   2);

  // ---- flash attention (WMMA) ----
  attn_kernel<<<dim3(NQP / 16 / 4, BQ * HEADS), 128, 0, stream>>>(qpool, kpool, vTp, o16);

  // ---- output projection ----
  dim3 gproj(DIMC / 256, BQ * NQP / 16);
  gemm_kernel<false, false><<<gproj, 128, 0, stream>>>(o16, wp16, projb, o2, ((_Float16*)nullptr), nullptr, BQ * NQP, DIMC, DIMC);

  // ---- max-pool skip + residual ----
  {
    size_t total = (size_t)BQ * NQ * DIMC;
    pool_add_kernel<<<dim3((unsigned)((total + 255) / 256)), 256, 0, stream>>>(x, o2, x2);
  }

  // ---- LN2 ----
  ln_kernel<<<dim3(BQ * NQP / 4), dim3(128), 0, stream>>>(x2, n2w, n2b, xn2, NQP, NQ);

  // ---- MLP: fc1 (GELU) then fc2 (+residual -> d_out) ----
  dim3 g1(HIDC / 256, BQ * NQP / 16);
  gemm_kernel<true, false><<<g1, 128, 0, stream>>>(xn2, w116, fc1b, (float*)nullptr, hbuf, nullptr, BQ * NQP, HIDC, DIMC);
  dim3 g2(DIMC / 256, BQ * NQP / 16);
  gemm_kernel<false, true><<<g2, 128, 0, stream>>>(hbuf, w216, fc2b, out, ((_Float16*)nullptr), x2, BQ * NQP, DIMC, HIDC);
}
